// BaselineAttention_44358422233487
// MI455X (gfx1250) — matrix operand order repaired and checked by reading, not compiled, not run on hardware
//
#include <hip/hip_runtime.h>
#include <math.h>

// ---------------------------------------------------------------- types
typedef _Float16 half_t;
typedef __attribute__((ext_vector_type(16))) _Float16 v16h;
typedef __attribute__((ext_vector_type(8)))  _Float16 v8h;
typedef __attribute__((ext_vector_type(8)))  float    v8f;

union HV16 { v16h v; half_t e[16]; v8h h[2]; };
union FV8  { v8f  v; float  f[8]; };

#define WMMA_F16(a, b, c) \
  __builtin_amdgcn_wmma_f32_16x16x32_f16(false, (a), false, (b), (short)0, (c), false, false)

// ---------------------------------------------------------------- problem dims
constexpr int kDim  = 1024;
constexpr int kH    = 16;
constexpr int kDh   = 64;        // per-head dim
constexpr int kB    = 4;
constexpr int kS    = 2048;
constexpr int kRows = kB * kS;   // 8192

// ---------------------------------------------------------------- frag loaders
// A-matrix 16x32 f16 (ISA 7.12.2): lane m = lane&15; lanes<16 hold K chunks
// {0..7, 16..23}, lanes>=16 hold {8..15, 24..31}.
__device__ inline v16h load_a_frag(const half_t* p, int ld, int row0, int k0) {
  int lane = threadIdx.x & 31;
  int m    = lane & 15;
  int kb   = (lane >> 4) << 3;   // 0 or 8
  const half_t* r = p + (size_t)(row0 + m) * ld + k0 + kb;
  HV16 a;
  a.h[0] = *(const v8h*)(r);
  a.h[1] = *(const v8h*)(r + 16);
  return a.v;
}

// B-matrix 32x16 f16: lane l holds column N=l&15, K in table order (hi-half offset 8).
__device__ inline v16h load_b_rows(const half_t* p, int ld, int k0, int n0) {
  int lane = threadIdx.x & 31;
  int kb   = (lane >> 4) << 3;
  const half_t* r = p + (size_t)k0 * ld + n0 + (lane & 15);
  HV16 b;
#pragma unroll
  for (int s = 0; s < 16; ++s)
    b.e[s] = r[(size_t)(kb + (s & 7) + ((s >> 3) << 4)) * ld];
  return b.v;
}

// ---------------------------------------------------------------- kernels
__global__ void cvt_f32_f16(const float* __restrict__ src,
                            half_t* __restrict__ dst, size_t n) {
  size_t i = (size_t)blockIdx.x * blockDim.x + threadIdx.x;
  if (i < n) dst[i] = (half_t)src[i];
}

// out[b,h,s,e] = x[b,s,:] @ W[h,:,e] + bias[h,e]
// One wave = 32(M) x 64(N) tile: two A-frags share every B-frag (8 WMMA / A-pair).
__global__ __launch_bounds__(32)
void proj_qkv(const half_t* __restrict__ x16, const half_t* __restrict__ W16,
              const float* __restrict__ bias, half_t* __restrict__ out) {
  int tile = blockIdx.x;            // 0..255 : 32-row tiles over [B*S]
  int h    = blockIdx.y;            // head
  int lane = threadIdx.x & 31;
  int n    = lane & 15;
  int hi   = (lane >> 4) << 3;
  const half_t* Wh = W16 + (size_t)h * kDim * kDh;

  FV8 acc[2][4];
#pragma unroll
  for (int nt = 0; nt < 4; ++nt) {
    float bv = bias[h * kDh + nt * 16 + n];
#pragma unroll
    for (int r = 0; r < 8; ++r) { acc[0][nt].f[r] = bv; acc[1][nt].f[r] = bv; }
  }

  int row0 = tile * 32;
  for (int k = 0; k < kDim; k += 32) {
    v16h a0 = load_a_frag(x16, kDim, row0,      k);
    v16h a1 = load_a_frag(x16, kDim, row0 + 16, k);
#pragma unroll
    for (int nt = 0; nt < 4; ++nt) {
      v16h b = load_b_rows(Wh, kDh, k, nt * 16);
      acc[0][nt].v = WMMA_F16(a0, b, acc[0][nt].v);
      acc[1][nt].v = WMMA_F16(a1, b, acc[1][nt].v);
    }
  }

  int b_ = row0 >> 11;              // /kS
  int s0 = row0 & (kS - 1);
  int bh = b_ * kH + h;
#pragma unroll
  for (int mt = 0; mt < 2; ++mt)
#pragma unroll
    for (int r = 0; r < 8; ++r) {
      int m = mt * 16 + r + hi;
      size_t o = ((size_t)bh * kS + s0 + m) * kDh;
#pragma unroll
      for (int nt = 0; nt < 4; ++nt)
        out[o + nt * 16 + n] = (half_t)acc[mt][nt].f[r];
    }
}

// Flash attention, one wave = 16 queries x 64 features for one (b,h).
// Scores computed transposed: S^T = K @ Q^T so K streams as contiguous A-frags
// and the only transposed gather (Q^T) happens once per wave. V tiles staged
// into LDS with the CDNA5 async-to-LDS engine (double-buffered).
__global__ __launch_bounds__(32)
void flash_attn(const half_t* __restrict__ Q16, const half_t* __restrict__ K16,
                const half_t* __restrict__ V16, half_t* __restrict__ O16) {
  __shared__ __align__(16) half_t P[16 * 32];        // P^T -> P staging
  __shared__ __align__(16) half_t Vt[2][32 * kDh];   // async V tiles (2 x 4KB)

  int qt   = blockIdx.x;            // query tile (16 rows)
  int bh   = blockIdx.y;            // b*kH + h
  int lane = threadIdx.x & 31;
  int m    = lane & 15;             // query column owned by this lane in S^T
  int hi   = (lane >> 4) << 3;

  const half_t* Qp = Q16 + (size_t)bh * kS * kDh;
  const half_t* Kp = K16 + (size_t)bh * kS * kDh;
  const half_t* Vp = V16 + (size_t)bh * kS * kDh;

  // B-fragments of Q^T (64 x 16), gathered once: B[k][n] = Q[q0+n][f0+k], n = lane&15, k in table order
  HV16 bq0, bq1;
#pragma unroll
  for (int n = 0; n < 16; ++n) {
    int kk = hi + (n & 7) + ((n >> 3) << 4);
    bq0.e[n] = Qp[(size_t)(qt * 16 + m) * kDh + kk];
    bq1.e[n] = Qp[(size_t)(qt * 16 + m) * kDh + 32 + kk];
  }

  FV8 o[4];
#pragma unroll
  for (int nt = 0; nt < 4; ++nt)
#pragma unroll
    for (int r = 0; r < 8; ++r) o[nt].f[r] = 0.f;
  float mrow = -3e38f, lrow = 0.f;   // stats for query m (per lane)

  const float scale = 0.125f;        // 1/sqrt(64)

  // LDS byte offsets of the two V buffers (generic-pointer low 32 bits == LDS offset)
  unsigned vb[2];
  vb[0] = (unsigned)(unsigned long long)(void*)&Vt[0][0];
  vb[1] = (unsigned)(unsigned long long)(void*)&Vt[1][0];

  for (int kc = 0; kc < kS; kc += 32) {
    int pb = (kc >> 5) & 1;

    // ---- async-stage V tile (32 rows x 64 f16 = 4KB, contiguous) into LDS
    {
      const char* vg = (const char*)(Vp + (size_t)kc * kDh) + lane * 16;
      unsigned    vl = vb[pb] + lane * 16;
#pragma unroll
      for (int i = 0; i < 8; ++i) {
        const void* ga = vg + i * 512;
        unsigned    la = vl + i * 512;
        asm volatile("global_load_async_to_lds_b128 %0, %1, off"
                     :: "v"(la), "v"(ga) : "memory");
      }
    }

    // ---- S^T tiles: [keys x queries] = K(32x64) @ Q^T(64x16)
    FV8 st0, st1;
#pragma unroll
    for (int r = 0; r < 8; ++r) { st0.f[r] = 0.f; st1.f[r] = 0.f; }
    {
      v16h ak0 = load_a_frag(Kp, kDh, kc, 0);
      v16h ak1 = load_a_frag(Kp, kDh, kc, 32);
      st0.v = WMMA_F16(ak0, bq0.v, st0.v);
      st0.v = WMMA_F16(ak1, bq1.v, st0.v);
      v16h ak2 = load_a_frag(Kp, kDh, kc + 16, 0);
      v16h ak3 = load_a_frag(Kp, kDh, kc + 16, 32);
      st1.v = WMMA_F16(ak2, bq0.v, st1.v);
      st1.v = WMMA_F16(ak3, bq1.v, st1.v);
    }

    // ---- online softmax for query m (lane dimension): 1 shuffle + 1 alpha exp
    float cm = -3e38f;
#pragma unroll
    for (int r = 0; r < 8; ++r) {
      st0.f[r] *= scale; st1.f[r] *= scale;
      cm = fmaxf(cm, fmaxf(st0.f[r], st1.f[r]));
    }
    cm = fmaxf(cm, __shfl_xor(cm, 16, 32));   // combine key halves
    float mnew  = fmaxf(mrow, cm);
    float alpha = __expf(mrow - mnew);
    mrow = mnew;

    float psum = 0.f;
#pragma unroll
    for (int r = 0; r < 8; ++r) {
      float p0 = __expf(st0.f[r] - mnew);
      float p1 = __expf(st1.f[r] - mnew);
      psum += p0 + p1;
      // P^T[t][m] -> P[m][t] row-major for the A-frag reload
      P[m * 32 + r + hi]      = (half_t)p0;   // keys kc   .. kc+15
      P[m * 32 + 16 + r + hi] = (half_t)p1;   // keys kc+16.. kc+31
    }
    lrow = lrow * alpha + psum;

    // broadcast alpha into O row space (row r+hi needs alpha of query r+hi)
    float aR[8];
#pragma unroll
    for (int r = 0; r < 8; ++r) aR[r] = __shfl(alpha, r + hi, 32);
#pragma unroll
    for (int nt = 0; nt < 4; ++nt)
#pragma unroll
      for (int r = 0; r < 8; ++r) o[nt].f[r] *= aR[r];

    asm volatile("s_wait_dscnt 0" ::: "memory");

    // ---- reload P as A-fragment (16x32)
    v16h ap;
    {
      HV16 t;
      const half_t* r = &P[m * 32 + hi];
      t.h[0] = *(const v8h*)(r);
      t.h[1] = *(const v8h*)(r + 16);
      ap = t.v;
    }

    // ---- P @ V from the async-staged LDS tile
    asm volatile("s_wait_asynccnt 0" ::: "memory");
    const half_t* vrow = &Vt[pb][m];   // lane&15 = feature column, keys in table order
#pragma unroll
  for (int nt = 0; nt < 4; ++nt) {
      HV16 bv;
#pragma unroll
      for (int s = 0; s < 16; ++s)
        bv.e[s] = vrow[(hi + (s & 7) + ((s >> 3) << 4)) * kDh + nt * 16];
      o[nt].v = WMMA_F16(ap, bv.v, o[nt].v);
    }
    asm volatile("s_wait_dscnt 0" ::: "memory");
  }

  // ---- normalize and store to concatenated-head layout [B,S,H*Dh]
  lrow += __shfl_xor(lrow, 16, 32);
  float inv = 1.0f / lrow;
  float invR[8];
#pragma unroll
  for (int r = 0; r < 8; ++r) invR[r] = __shfl(inv, r + hi, 32);

  int b_ = bh >> 4, h_ = bh & 15;
#pragma unroll
  for (int r = 0; r < 8; ++r) {
    int mm = r + hi;
    size_t s   = (size_t)qt * 16 + mm;
    size_t off = ((size_t)b_ * kS + s) * kDim + (size_t)h_ * kDh;
#pragma unroll
    for (int nt = 0; nt < 4; ++nt)
      O16[off + nt * 16 + m] = (half_t)(o[nt].f[r] * invR[r]);
  }
}

// out[r, c] = O16[r, :] @ Wo16[:, c] + bo[c]   (32x64 tile, fp32 epilogue)
__global__ __launch_bounds__(32)
void out_proj(const half_t* __restrict__ O16, const half_t* __restrict__ Wo16,
              const float* __restrict__ bo, float* __restrict__ out) {
  int tile = blockIdx.x;            // 256 row tiles of 32
  int ct   = blockIdx.y;            // 16 column tiles of 64
  int lane = threadIdx.x & 31;
  int n    = lane & 15;
  int hi   = (lane >> 4) << 3;

  FV8 acc[2][4];
#pragma unroll
  for (int nt = 0; nt < 4; ++nt) {
    float bv = bo[ct * 64 + nt * 16 + n];
#pragma unroll
    for (int r = 0; r < 8; ++r) { acc[0][nt].f[r] = bv; acc[1][nt].f[r] = bv; }
  }

  int row0 = tile * 32;
  for (int k = 0; k < kDim; k += 32) {
    v16h a0 = load_a_frag(O16, kDim, row0,      k);
    v16h a1 = load_a_frag(O16, kDim, row0 + 16, k);
#pragma unroll
    for (int nt = 0; nt < 4; ++nt) {
      v16h b = load_b_rows(Wo16, kDim, k, ct * 64 + nt * 16);
      acc[0][nt].v = WMMA_F16(a0, b, acc[0][nt].v);
      acc[1][nt].v = WMMA_F16(a1, b, acc[1][nt].v);
    }
  }

#pragma unroll
  for (int mt = 0; mt < 2; ++mt)
#pragma unroll
    for (int r = 0; r < 8; ++r) {
      int mm = mt * 16 + r + hi;
      size_t off = (size_t)(row0 + mm) * kDim + (size_t)ct * 64;
#pragma unroll
      for (int nt = 0; nt < 4; ++nt)
        out[off + nt * 16 + n] = acc[mt][nt].f[r];
    }
}

// ---------------------------------------------------------------- host entry
extern "C" void kernel_launch(void* const* d_in, const int* in_sizes, int n_in,
                              void* d_out, int out_size, void* d_ws, size_t ws_size,
                              hipStream_t stream) {
  (void)in_sizes; (void)n_in; (void)out_size; (void)ws_size;

  const float* x  = (const float*)d_in[0];
  const float* Wq = (const float*)d_in[1];
  const float* Wk = (const float*)d_in[2];
  const float* Wv = (const float*)d_in[3];
  const float* bq = (const float*)d_in[4];
  const float* bk = (const float*)d_in[5];
  const float* bv = (const float*)d_in[6];
  const float* Wo = (const float*)d_in[7];
  const float* bo = (const float*)d_in[8];
  float* out = (float*)d_out;

  const size_t NX = (size_t)kRows * kDim;     // 8M elems
  const size_t NW = (size_t)kH * kDim * kDh;  // 1M elems
  half_t* ws   = (half_t*)d_ws;
  half_t* x16  = ws;
  half_t* Wq16 = x16  + NX;
  half_t* Wk16 = Wq16 + NW;
  half_t* Wv16 = Wk16 + NW;
  half_t* Wo16 = Wv16 + NW;
  half_t* Q16  = Wo16 + NW;                   // [B,H,S,Dh]
  half_t* K16  = Q16  + NX;
  half_t* V16  = K16  + NX;
  half_t* O16  = V16  + NX;                   // [B,S,H*Dh]

  {
    int t = 256;
    cvt_f32_f16<<<dim3((unsigned)((NX + t - 1) / t)), dim3(t), 0, stream>>>(x,  x16,  NX);
    cvt_f32_f16<<<dim3((unsigned)((NW + t - 1) / t)), dim3(t), 0, stream>>>(Wq, Wq16, NW);
    cvt_f32_f16<<<dim3((unsigned)((NW + t - 1) / t)), dim3(t), 0, stream>>>(Wk, Wk16, NW);
    cvt_f32_f16<<<dim3((unsigned)((NW + t - 1) / t)), dim3(t), 0, stream>>>(Wv, Wv16, NW);
    cvt_f32_f16<<<dim3((unsigned)((NW + t - 1) / t)), dim3(t), 0, stream>>>(Wo, Wo16, NW);
  }

  dim3 gp(kRows / 32, kH);
  proj_qkv<<<gp, 32, 0, stream>>>(x16, Wq16, bq, Q16);
  proj_qkv<<<gp, 32, 0, stream>>>(x16, Wk16, bk, K16);
  proj_qkv<<<gp, 32, 0, stream>>>(x16, Wv16, bv, V16);

  flash_attn<<<dim3(kS / 16, kB * kH), 32, 0, stream>>>(Q16, K16, V16, O16);

  out_proj<<<dim3(kRows / 32, kDim / 64), 32, 0, stream>>>(O16, Wo16, bo, out);
}
